// CausalSelfAttention_1322849927676
// MI455X (gfx1250) — matrix operand order repaired and checked by reading, not compiled, not run on hardware
//
#include <hip/hip_runtime.h>

// ---------------- problem constants ----------------
constexpr int Bc = 4;
constexpr int Tc = 2048;
constexpr int Cc = 1024;
constexpr int Hc = 16;
constexpr int Dc = 64;          // head dim
constexpr int NTOK = Bc * Tc;   // 8192 tokens

typedef __attribute__((ext_vector_type(16))) __bf16          v16bf;
typedef __attribute__((ext_vector_type(8)))  float           v8f;
typedef __attribute__((ext_vector_type(16))) unsigned short  v16u;
typedef __attribute__((ext_vector_type(8)))  unsigned short  v8u;

// ---------------- helpers ----------------
__device__ __forceinline__ unsigned short f2bf(float f) {
    unsigned int u = __builtin_bit_cast(unsigned int, f);
    unsigned int r = u + 0x7FFFu + ((u >> 16) & 1u);   // round-to-nearest-even
    return (unsigned short)(r >> 16);
}

__device__ __forceinline__ v8f zero8() {
    v8f z = {0.f, 0.f, 0.f, 0.f, 0.f, 0.f, 0.f, 0.f};
    return z;
}

// D = A(16x32 bf16) x B(32x16 bf16) + C(16x16 f32)
__device__ __forceinline__ v8f wmma_bf16(v16bf a, v16bf b, v8f c) {
    return __builtin_amdgcn_wmma_f32_16x16x32_bf16(
        /*neg_a=*/false, a, /*neg_b=*/false, b,
        /*c_mod=*/(short)0, c, /*reuse_a=*/false, /*reuse_b=*/false);
}

// A-fragment: lane holds row (lane&15); elements e: k = kbase + e%8 + (e/8)*16,
// kbase = (lane<16)?0:8  ->  two contiguous 16-byte chunks at p and p+16 elems.
__device__ __forceinline__ v16bf load_a_frag(const unsigned short* p) {
    union { v16u v; v8u h[2]; } u;
    u.h[0] = *(const v8u*)(p);
    u.h[1] = *(const v8u*)(p + 16);
    return __builtin_bit_cast(v16bf, u.v);
}

// B-fragment: lane holds column (lane&15); elements e: k = kbase + e%8 + (e/8)*16,
// kbase = (lane<16)?0:8  ->  two contiguous 16-byte chunks at p and p+16 elems.
__device__ __forceinline__ v16bf load_b_frag(const unsigned short* p) {
    union { v16u v; v8u h[2]; } u;
    u.h[0] = *(const v8u*)(p);
    u.h[1] = *(const v8u*)(p + 16);
    return __builtin_bit_cast(v16bf, u.v);
}

// ---------------- kernel 1: fp32 -> bf16 conversion ----------------
__global__ __launch_bounds__(256) void cvt_bf16(const float* __restrict__ in,
                                                unsigned short* __restrict__ out,
                                                int n) {
    int i = (blockIdx.x * 256 + threadIdx.x) * 4;
    if (i >= n) return;
    float4 f = *(const float4*)(in + i);
    unsigned long long p =
        (unsigned long long)f2bf(f.x)
      | ((unsigned long long)f2bf(f.y) << 16)
      | ((unsigned long long)f2bf(f.z) << 32)
      | ((unsigned long long)f2bf(f.w) << 48);
    *(unsigned long long*)(out + i) = p;
}

// ---------------- kernel 2: fused QKV projection ----------------
// q,k,v = x @ W^T + b.  Each wave computes a 32x64 macro-tile (2x4 register
// blocking): 6 fragment loads feed 8 WMMAs per k-step.
// q,k stored [B,H,T,D] bf16; v stored transposed [B,H,D,T].
__global__ __launch_bounds__(256) void qkv_gemm(
    const unsigned short* __restrict__ xb,
    const unsigned short* __restrict__ wqb,
    const unsigned short* __restrict__ wkb,
    const unsigned short* __restrict__ wvb,
    const float* __restrict__ bq, const float* __restrict__ bk,
    const float* __restrict__ bv,
    unsigned short* __restrict__ qh, unsigned short* __restrict__ kh,
    unsigned short* __restrict__ vt) {
    const int lane = threadIdx.x & 31;
    const int w    = threadIdx.x >> 5;
    const int gw   = blockIdx.x * 8 + w;

    constexpr int MT    = NTOK / 32;          // 256 macro-rows
    constexpr int NT    = Cc / 64;            // 16 macro-cols
    constexpr int TILES = MT * NT;            // 4096 per matrix

    const int which = gw / TILES;             // 0=q 1=k 2=v
    const int rem   = gw % TILES;
    const int mt    = rem / NT;
    const int nt    = rem % NT;

    const unsigned short* W    = (which == 0) ? wqb : ((which == 1) ? wkb : wvb);
    const float*          bias = (which == 0) ? bq  : ((which == 1) ? bk  : bv);

    const int akb = (lane < 16) ? 0 : 8;
    const int bkb = (lane < 16) ? 0 : 8;
    const unsigned short* aptr0 = xb + (size_t)(mt * 32 + (lane & 15)) * Cc + akb;
    const unsigned short* aptr1 = aptr0 + (size_t)16 * Cc;
    const unsigned short* bptr  = W  + (size_t)(nt * 64 + (lane & 15)) * Cc + bkb;

    v8f acc[2][4];
    #pragma unroll
    for (int mi = 0; mi < 2; ++mi)
        #pragma unroll
        for (int ni = 0; ni < 4; ++ni) acc[mi][ni] = zero8();

    #pragma unroll 2
    for (int kk = 0; kk < Cc; kk += 32) {
        const v16bf a0 = load_a_frag(aptr0 + kk);
        const v16bf a1 = load_a_frag(aptr1 + kk);
        const v16bf b0 = load_b_frag(bptr + kk);
        const v16bf b1 = load_b_frag(bptr + (size_t)16 * Cc + kk);
        const v16bf b2 = load_b_frag(bptr + (size_t)32 * Cc + kk);
        const v16bf b3 = load_b_frag(bptr + (size_t)48 * Cc + kk);
        acc[0][0] = wmma_bf16(a0, b0, acc[0][0]);
        acc[0][1] = wmma_bf16(a0, b1, acc[0][1]);
        acc[0][2] = wmma_bf16(a0, b2, acc[0][2]);
        acc[0][3] = wmma_bf16(a0, b3, acc[0][3]);
        acc[1][0] = wmma_bf16(a1, b0, acc[1][0]);
        acc[1][1] = wmma_bf16(a1, b1, acc[1][1]);
        acc[1][2] = wmma_bf16(a1, b2, acc[1][2]);
        acc[1][3] = wmma_bf16(a1, b3, acc[1][3]);
    }

    #pragma unroll
    for (int ni = 0; ni < 4; ++ni) {
        const int   col = nt * 64 + ni * 16 + (lane & 15);  // lane holds column N
        const float bb  = bias[col];
        const int   h   = col / Dc;
        const int   dd  = col % Dc;
        #pragma unroll
        for (int mi = 0; mi < 2; ++mi) {
            #pragma unroll
            for (int v = 0; v < 8; ++v) {
                const int m  = mt * 32 + mi * 16 + v + 8 * (lane >> 4);
                const int bi = m / Tc;
                const int t  = m % Tc;
                const unsigned short bf = f2bf(acc[mi][ni][v] + bb);
                const size_t hb = (size_t)(bi * Hc + h);
                if (which == 0)      qh[(hb * Tc + t) * Dc + dd] = bf;
                else if (which == 1) kh[(hb * Tc + t) * Dc + dd] = bf;
                else                 vt[(hb * Dc + dd) * Tc + t] = bf;   // transposed
            }
        }
    }
}

// ---------------- kernel 3: causal flash attention ----------------
// One wave = 16 query rows. S^T = K @ Q^T so softmax reductions are in-lane,
// and exp(P) lands directly in the P@V A-fragment register pattern.
__global__ __launch_bounds__(256) void attn(
    const unsigned short* __restrict__ qh, const unsigned short* __restrict__ kh,
    const unsigned short* __restrict__ vt, unsigned short* __restrict__ yb) {
    const int lane = threadIdx.x & 31;
    const int w    = threadIdx.x >> 5;

    constexpr int QG = Tc / 128;             // 16 q-groups of 8 waves
    const int bh    = blockIdx.x / QG;       // batch*head
    const int qg    = blockIdx.x % QG;
    const int b     = bh / Hc;
    const int h     = bh % Hc;
    const int qbase = (qg * 8 + w) * 16;

    const unsigned short* qp = qh + (size_t)bh * Tc * Dc;
    const unsigned short* kp = kh + (size_t)bh * Tc * Dc;
    const unsigned short* vp = vt + (size_t)bh * Dc * Tc;

    const int akb = (lane < 16) ? 0 : 8;
    const int bkb = (lane < 16) ? 0 : 8;

    // Q^T B-fragments (column = query), persistent across the key loop
    const v16bf qf0 = load_b_frag(qp + (size_t)(qbase + (lane & 15)) * Dc + 0  + bkb);
    const v16bf qf1 = load_b_frag(qp + (size_t)(qbase + (lane & 15)) * Dc + 32 + bkb);

    v8f o0 = zero8(), o1 = zero8(), o2 = zero8(), o3 = zero8();
    float mrun = -__builtin_inff();
    float lrun = 0.f;
    const float scale = 0.125f;              // 1/sqrt(64)
    const int   qrow  = qbase + (lane & 15);
    const int   kend  = qbase + 16;          // causal bound for this wave
    const int   qloc  = 8 * (lane >> 4);

    for (int kb = 0; kb < kend; kb += 32) {
        __builtin_prefetch(kp + (size_t)(kb + 32) * Dc, 0, 1);

        // S^T tiles: rows = keys, cols = queries
        v8f s0 = zero8(), s1 = zero8();
        {
            const unsigned short* ka0 = kp + (size_t)(kb + (lane & 15)) * Dc + akb;
            s0 = wmma_bf16(load_a_frag(ka0),      qf0, s0);
            s0 = wmma_bf16(load_a_frag(ka0 + 32), qf1, s0);
            const unsigned short* ka1 = kp + (size_t)(kb + 16 + (lane & 15)) * Dc + akb;
            s1 = wmma_bf16(load_a_frag(ka1),      qf0, s1);
            s1 = wmma_bf16(load_a_frag(ka1 + 32), qf1, s1);
        }

        // scale + causal mask + per-query running max (16 in-lane + xor16)
        const int kr0 = kb + qloc;
        const int kr1 = kb + 16 + qloc;
        float lm = -__builtin_inff();
        #pragma unroll
        for (int v = 0; v < 8; ++v) {
            float a = s0[v] * scale; if (kr0 + v > qrow) a = -__builtin_inff();
            s0[v] = a; lm = fmaxf(lm, a);
            float c = s1[v] * scale; if (kr1 + v > qrow) c = -__builtin_inff();
            s1[v] = c; lm = fmaxf(lm, c);
        }
        lm = fmaxf(lm, __shfl_xor(lm, 16, 32));
        const float mnew  = fmaxf(mrun, lm);
        const float alpha = __expf(mrun - mnew);

        // exp + row sum; pack P into A-fragment order (e<8 -> s0, e>=8 -> s1)
        float ls = 0.f;
        unsigned short pu[16];
        #pragma unroll
        for (int v = 0; v < 8; ++v) {
            float p0 = __expf(s0[v] - mnew); ls += p0; pu[v]     = f2bf(p0);
            float p1 = __expf(s1[v] - mnew); ls += p1; pu[8 + v] = f2bf(p1);
        }
        ls  += __shfl_xor(ls, 16, 32);
        lrun = lrun * alpha + ls;
        mrun = mnew;

        // rescale O rows (O row = query = v + 8*(lane/16); alpha lives at lane==query)
        #pragma unroll
        for (int v = 0; v < 8; ++v) {
            const float av = __shfl(alpha, qloc + v, 32);
            o0[v] *= av; o1[v] *= av; o2[v] *= av; o3[v] *= av;
        }

        v16bf pa;
        {
            union { v16u u; unsigned short s[16]; } cv;
            #pragma unroll
            for (int e = 0; e < 16; ++e) cv.s[e] = pu[e];
            pa = __builtin_bit_cast(v16bf, cv.u);
        }

        // O += P @ V  (V stored [D,T]: B-fragment is two 16B loads along T)
        o0 = wmma_bf16(pa, load_b_frag(vp + (size_t)( 0 + (lane & 15)) * Tc + kb + bkb), o0);
        o1 = wmma_bf16(pa, load_b_frag(vp + (size_t)(16 + (lane & 15)) * Tc + kb + bkb), o1);
        o2 = wmma_bf16(pa, load_b_frag(vp + (size_t)(32 + (lane & 15)) * Tc + kb + bkb), o2);
        o3 = wmma_bf16(pa, load_b_frag(vp + (size_t)(48 + (lane & 15)) * Tc + kb + bkb), o3);
    }

    // normalize and write token-major bf16 [NTOK, C]
    const float linv = 1.f / lrun;
    #pragma unroll
    for (int v = 0; v < 8; ++v) {
        const float rn = __shfl(linv, qloc + v, 32);
        const int   q  = qbase + qloc + v;
        const size_t base = (size_t)(b * Tc + q) * Cc + h * Dc + (lane & 15);
        yb[base +  0] = f2bf(o0[v] * rn);
        yb[base + 16] = f2bf(o1[v] * rn);
        yb[base + 32] = f2bf(o2[v] * rn);
        yb[base + 48] = f2bf(o3[v] * rn);
    }
}

// ---------------- kernel 4: output projection (fp32 out) ----------------
// Same 2x4 register blocking as qkv_gemm.
__global__ __launch_bounds__(256) void out_proj(
    const unsigned short* __restrict__ yb,
    const unsigned short* __restrict__ wob,
    const float* __restrict__ bo,
    float* __restrict__ out) {
    const int lane = threadIdx.x & 31;
    const int w    = threadIdx.x >> 5;
    const int gw   = blockIdx.x * 8 + w;

    constexpr int NT = Cc / 64;   // 16 macro-cols
    const int mt = gw / NT;
    const int nt = gw % NT;

    const int akb = (lane < 16) ? 0 : 8;
    const int bkb = (lane < 16) ? 0 : 8;
    const unsigned short* aptr0 = yb  + (size_t)(mt * 32 + (lane & 15)) * Cc + akb;
    const unsigned short* aptr1 = aptr0 + (size_t)16 * Cc;
    const unsigned short* bptr  = wob + (size_t)(nt * 64 + (lane & 15)) * Cc + bkb;

    v8f acc[2][4];
    #pragma unroll
    for (int mi = 0; mi < 2; ++mi)
        #pragma unroll
        for (int ni = 0; ni < 4; ++ni) acc[mi][ni] = zero8();

    #pragma unroll 2
    for (int kk = 0; kk < Cc; kk += 32) {
        const v16bf a0 = load_a_frag(aptr0 + kk);
        const v16bf a1 = load_a_frag(aptr1 + kk);
        const v16bf b0 = load_b_frag(bptr + kk);
        const v16bf b1 = load_b_frag(bptr + (size_t)16 * Cc + kk);
        const v16bf b2 = load_b_frag(bptr + (size_t)32 * Cc + kk);
        const v16bf b3 = load_b_frag(bptr + (size_t)48 * Cc + kk);
        acc[0][0] = wmma_bf16(a0, b0, acc[0][0]);
        acc[0][1] = wmma_bf16(a0, b1, acc[0][1]);
        acc[0][2] = wmma_bf16(a0, b2, acc[0][2]);
        acc[0][3] = wmma_bf16(a0, b3, acc[0][3]);
        acc[1][0] = wmma_bf16(a1, b0, acc[1][0]);
        acc[1][1] = wmma_bf16(a1, b1, acc[1][1]);
        acc[1][2] = wmma_bf16(a1, b2, acc[1][2]);
        acc[1][3] = wmma_bf16(a1, b3, acc[1][3]);
    }

    #pragma unroll
    for (int ni = 0; ni < 4; ++ni) {
        const int   col = nt * 64 + ni * 16 + (lane & 15);
        const float bb  = bo[col];
        #pragma unroll
        for (int mi = 0; mi < 2; ++mi) {
            #pragma unroll
            for (int v = 0; v < 8; ++v) {
                const int m = mt * 32 + mi * 16 + v + 8 * (lane >> 4);
                out[(size_t)m * Cc + col] = acc[mi][ni][v] + bb;
            }
        }
    }
}

// ---------------- launch ----------------
extern "C" void kernel_launch(void* const* d_in, const int* in_sizes, int n_in,
                              void* d_out, int out_size, void* d_ws, size_t ws_size,
                              hipStream_t stream) {
    const float* x  = (const float*)d_in[0];
    const float* Wq = (const float*)d_in[1];
    const float* bq = (const float*)d_in[2];
    const float* Wk = (const float*)d_in[3];
    const float* bk = (const float*)d_in[4];
    const float* Wv = (const float*)d_in[5];
    const float* bv = (const float*)d_in[6];
    const float* Wo = (const float*)d_in[7];
    const float* bo = (const float*)d_in[8];
    float* out = (float*)d_out;

    // workspace layout (bf16 buffers), ~88 MB total
    char* ws = (char*)d_ws;
    const size_t SZ_X  = (size_t)NTOK * Cc * 2;   // 16 MB
    const size_t SZ_W  = (size_t)Cc * Cc * 2;     //  2 MB
    const size_t SZ_H  = (size_t)Bc * Hc * Tc * Dc * 2; // 16 MB
    unsigned short* xb  = (unsigned short*)(ws);
    unsigned short* wqb = (unsigned short*)(ws + SZ_X);
    unsigned short* wkb = (unsigned short*)(ws + SZ_X + SZ_W);
    unsigned short* wvb = (unsigned short*)(ws + SZ_X + 2 * SZ_W);
    unsigned short* wob = (unsigned short*)(ws + SZ_X + 3 * SZ_W);
    unsigned short* qhb = (unsigned short*)(ws + SZ_X + 4 * SZ_W);
    unsigned short* khb = (unsigned short*)(ws + SZ_X + 4 * SZ_W + SZ_H);
    unsigned short* vtb = (unsigned short*)(ws + SZ_X + 4 * SZ_W + 2 * SZ_H);
    unsigned short* ybb = (unsigned short*)(ws + SZ_X + 4 * SZ_W + 3 * SZ_H);

    // 1) fp32 -> bf16
    cvt_bf16<<<(NTOK * Cc) / 1024, 256, 0, stream>>>(x,  xb,  NTOK * Cc);
    cvt_bf16<<<(Cc * Cc) / 1024,   256, 0, stream>>>(Wq, wqb, Cc * Cc);
    cvt_bf16<<<(Cc * Cc) / 1024,   256, 0, stream>>>(Wk, wkb, Cc * Cc);
    cvt_bf16<<<(Cc * Cc) / 1024,   256, 0, stream>>>(Wv, wvb, Cc * Cc);
    cvt_bf16<<<(Cc * Cc) / 1024,   256, 0, stream>>>(Wo, wob, Cc * Cc);

    // 2) fused QKV projection: 3 * 256 * 16 macro-tiles, 8 waves/block
    qkv_gemm<<<(3 * (NTOK / 32) * (Cc / 64)) / 8, 256, 0, stream>>>(
        xb, wqb, wkb, wvb, bq, bk, bv, qhb, khb, vtb);

    // 3) causal flash attention: B*H*(T/128) blocks of 8 waves
    attn<<<Bc * Hc * (Tc / 128), 256, 0, stream>>>(qhb, khb, vtb, ybb);

    // 4) output projection: 256 * 16 macro-tiles, 8 waves/block
    out_proj<<<((NTOK / 32) * (Cc / 64)) / 8, 256, 0, stream>>>(ybb, wob, bo, out);
}
